// TransformerNetwork_44770739093765
// MI455X (gfx1250) — matrix operand order repaired and checked by reading, not compiled, not run on hardware
//
#include <hip/hip_runtime.h>
#include <math.h>
#include <stdint.h>

// ---------------------------------------------------------------------------
// Transformer forward for MI455X (gfx1250, wave32, WMMA 16x16x32 f16->f32)
// B=16 T=256 D=768 H=12 DH=64 F=3072 L=6 V=1024
// ---------------------------------------------------------------------------

#define B_ 16
#define T_ 256
#define D_ 768
#define H_ 12
#define DH_ 64
#define F_ 3072
#define L_ 6
#define V_ 1024

#define KSPLIT 48          // head GEMM split-K chunks (196608 / 48 = 4096)
#define KCHUNK 4096

typedef __attribute__((ext_vector_type(16))) _Float16 v16h;
typedef __attribute__((ext_vector_type(8)))  _Float16 v8h;
typedef __attribute__((ext_vector_type(4)))  _Float16 v4h;
typedef __attribute__((ext_vector_type(8)))  float    v8f;
typedef __attribute__((ext_vector_type(4)))  float    v4f;

// GEMM tiling: block tile 128x128, staged K depth 64 (two WMMA K-steps/stage)
#define BM 128
#define BN 128
#define BKT 64
#define BKP 72   // padded K-stride in halves; 144B rows keep 16B alignment

// ---------------------------------------------------------------------------
// Generic WMMA GEMM: C[z] = A[z] @ B[z] (+bias) (+relu)
// A: [M,K] row-major (lda), B: [K,N] row-major (ldb) or [N,K] if TRANSB,
// C: [M,N] (ldc). Two batch dims: z = z1*Z2 + z2 with independent strides.
// FULL=1: M%BM==0, N%BN==0, K%BKT==0 -> all guards compiled out.
// NT=1: B-matrix loads are nontemporal (single-use streaming weights).
// fp32 in memory, f16 in LDS/WMMA, fp32 accumulate.
// ---------------------------------------------------------------------------
template<int TRANSB, int RELU, int FULL, int NT>
__global__ __launch_bounds__(256) void gemm_wmma_kernel(
    const float* __restrict__ A, const float* __restrict__ Bm,
    const float* __restrict__ bias, float* __restrict__ C,
    int M, int N, int K, int lda, int ldb, int ldc,
    long long aS1, long long aS2, long long bS1, long long bS2,
    long long cS1, long long cS2, int Z2)
{
    __shared__ __align__(16) _Float16 As[BM * BKP];
    __shared__ __align__(16) _Float16 Bs[BN * BKP];

    const int z  = blockIdx.z;
    const int z1 = z / Z2;
    const int z2 = z - z1 * Z2;
    A  += z1 * aS1 + z2 * aS2;
    Bm += z1 * bS1 + z2 * bS2;
    C  += z1 * cS1 + z2 * cS2;

    const int tid  = threadIdx.x;
    const int lane = tid & 31;
    const int wave = tid >> 5;
    const int wm   = wave >> 1;        // 0..3  -> 32 rows each
    const int wn   = wave & 1;         // 0..1  -> 64 cols each
    const int lr   = lane & 15;
    const int hsel = (lane >> 4) & 1;  // half-wave select
    const int rowBase = blockIdx.y * BM;
    const int colBase = blockIdx.x * BN;

    v8f acc[2][4];
#pragma unroll
    for (int mi = 0; mi < 2; ++mi)
#pragma unroll
        for (int ni = 0; ni < 4; ++ni)
            acc[mi][ni] = v8f{0.f,0.f,0.f,0.f,0.f,0.f,0.f,0.f};

    const int kTiles = (K + BKT - 1) / BKT;
    for (int kt = 0; kt < kTiles; ++kt) {
        const int k0 = kt * BKT;

        // ---- stage A tile: 128x64 f32 -> f16 LDS (8 v4f loads, b64 stores) ----
#pragma unroll
        for (int i = 0; i < 8; ++i) {
            const int g   = tid + i * 256;
            const int row = g >> 4;            // 16 v4f groups per row
            const int kq  = (g & 15) << 2;
            float v0 = 0.f, v1 = 0.f, v2 = 0.f, v3 = 0.f;
            if (FULL) {
                const float* p = A + (long long)(rowBase + row) * lda + k0 + kq;
                const v4f f = *(const v4f*)p;
                v0 = f[0]; v1 = f[1]; v2 = f[2]; v3 = f[3];
                __builtin_prefetch(p + BKT, 0, 3);   // next K tile
            } else {
                const int gr = rowBase + row;
                if (gr < M) {
                    const float* p = A + (long long)gr * lda + k0 + kq;
                    if (k0 + kq + 4 <= K) {
                        const v4f f = *(const v4f*)p;
                        v0 = f[0]; v1 = f[1]; v2 = f[2]; v3 = f[3];
                    } else {
                        const int rem = K - (k0 + kq);
                        if (rem > 0) v0 = p[0];
                        if (rem > 1) v1 = p[1];
                        if (rem > 2) v2 = p[2];
                    }
                }
            }
            const v4h pack = {(_Float16)v0, (_Float16)v1, (_Float16)v2, (_Float16)v3};
            *(v4h*)&As[row * BKP + kq] = pack;
        }

        // ---- stage B tile into Bs[n][k] layout ----
        if (TRANSB) {
            // B is [N,K] row-major: contiguous along k, same pattern as A.
#pragma unroll
            for (int i = 0; i < 8; ++i) {
                const int g  = tid + i * 256;
                const int n  = g >> 4;
                const int kq = (g & 15) << 2;
                float v0 = 0.f, v1 = 0.f, v2 = 0.f, v3 = 0.f;
                if (FULL) {
                    const float* p = Bm + (long long)(colBase + n) * ldb + k0 + kq;
                    const v4f f = NT ? __builtin_nontemporal_load((const v4f*)p)
                                     : *(const v4f*)p;
                    v0 = f[0]; v1 = f[1]; v2 = f[2]; v3 = f[3];
                    __builtin_prefetch(p + BKT, 0, 3);
                } else {
                    const int gn = colBase + n;
                    if (gn < N) {
                        const float* p = Bm + (long long)gn * ldb + k0 + kq;
                        if (k0 + kq + 4 <= K) {
                            const v4f f = NT ? __builtin_nontemporal_load((const v4f*)p)
                                             : *(const v4f*)p;
                            v0 = f[0]; v1 = f[1]; v2 = f[2]; v3 = f[3];
                        } else {
                            const int rem = K - (k0 + kq);
                            if (rem > 0) v0 = p[0];
                            if (rem > 1) v1 = p[1];
                            if (rem > 2) v2 = p[2];
                        }
                    }
                }
                const v4h pack = {(_Float16)v0, (_Float16)v1, (_Float16)v2, (_Float16)v3};
                *(v4h*)&Bs[n * BKP + kq] = pack;
            }
        } else {
            // B is [K,N] row-major: 4x4 micro-tile transpose in registers.
            // Loads: coalesced v4f along n (wave covers full 128-col row).
            // Stores: contiguous-in-k v4h (b64) into Bs[n][k].
#pragma unroll
            for (int i = 0; i < 2; ++i) {
                const int mt    = tid + i * 256;     // 512 micro-tiles (16k x 32n)
                const int ntile = (mt & 31) << 2;    // 0..124
                const int ktile = (mt >> 5) << 2;    // 0..60
                float rr[4][4];
#pragma unroll
                for (int j = 0; j < 4; ++j) {
                    rr[j][0] = 0.f; rr[j][1] = 0.f; rr[j][2] = 0.f; rr[j][3] = 0.f;
                    if (FULL) {
                        const float* p = Bm + (long long)(k0 + ktile + j) * ldb
                                            + colBase + ntile;
                        const v4f f = NT ? __builtin_nontemporal_load((const v4f*)p)
                                         : *(const v4f*)p;
                        rr[j][0] = f[0]; rr[j][1] = f[1]; rr[j][2] = f[2]; rr[j][3] = f[3];
                        if (j == 0) __builtin_prefetch(p + (long long)BKT * ldb, 0, 3);
                    } else {
                        const int gk = k0 + ktile + j;
                        if (gk < K) {
                            const float* p = Bm + (long long)gk * ldb + colBase + ntile;
                            if (colBase + ntile + 4 <= N) {
                                const v4f f = NT ? __builtin_nontemporal_load((const v4f*)p)
                                                 : *(const v4f*)p;
                                rr[j][0] = f[0]; rr[j][1] = f[1];
                                rr[j][2] = f[2]; rr[j][3] = f[3];
                            } else {
                                const int rem = N - (colBase + ntile);
                                if (rem > 0) rr[j][0] = p[0];
                                if (rem > 1) rr[j][1] = p[1];
                                if (rem > 2) rr[j][2] = p[2];
                            }
                        }
                    }
                }
#pragma unroll
                for (int c = 0; c < 4; ++c) {
                    const v4h pack = {(_Float16)rr[0][c], (_Float16)rr[1][c],
                                      (_Float16)rr[2][c], (_Float16)rr[3][c]};
                    *(v4h*)&Bs[(ntile + c) * BKP + ktile] = pack;
                }
            }
        }
        __syncthreads();

        // ---- 2 K-steps x 8 WMMAs per staged tile ----
        // A layout (16-bit 16x32, ISA 7.12.2): lanes 0-15 hold K {0-7,16-23},
        // lanes 16-31 hold K {8-15,24-31}  -> two b128 loads at 8*hsel / 16+8*hsel.
        // B layout: same K order as A -> lanes hold K {8*hsel..+7, 16+8*hsel..+7}.
#pragma unroll
        for (int ks = 0; ks < 2; ++ks) {
            const int koff = ks * 32;
            v16h af[2], bf[4];
            const int akb = koff + (hsel ? 8 : 0);
#pragma unroll
            for (int mi = 0; mi < 2; ++mi) {
                const int base = (wm * 32 + mi * 16 + lr) * BKP;
                v8h lo = *(const v8h*)&As[base + akb];
                v8h hi = *(const v8h*)&As[base + akb + 16];
                af[mi] = __builtin_shufflevector(lo, hi,
                    0,1,2,3,4,5,6,7,8,9,10,11,12,13,14,15);
            }
            const int bkb = koff + (hsel ? 8 : 0);
#pragma unroll
            for (int ni = 0; ni < 4; ++ni) {
                const int base = (wn * 64 + ni * 16 + lr) * BKP;
                v8h lo = *(const v8h*)&Bs[base + bkb];
                v8h hi = *(const v8h*)&Bs[base + bkb + 16];
                bf[ni] = __builtin_shufflevector(lo, hi,
                    0,1,2,3,4,5,6,7,8,9,10,11,12,13,14,15);
            }
#pragma unroll
            for (int mi = 0; mi < 2; ++mi)
#pragma unroll
                for (int ni = 0; ni < 4; ++ni)
                    acc[mi][ni] = __builtin_amdgcn_wmma_f32_16x16x32_f16(
                        false, af[mi], false, bf[ni],
                        (short)0, acc[mi][ni], false, false);
        }
        __syncthreads();
    }

    // ---- epilogue: D layout = lanes 0-15 M=r, lanes 16-31 M=r+8, N=lane%16 ----
#pragma unroll
    for (int mi = 0; mi < 2; ++mi) {
#pragma unroll
        for (int ni = 0; ni < 4; ++ni) {
            const int col = colBase + wn * 64 + ni * 16 + lr;
            const bool cok = FULL ? true : (col < N);
            const float bv = (bias != nullptr && cok) ? bias[col] : 0.f;
#pragma unroll
            for (int r = 0; r < 8; ++r) {
                const int row = rowBase + wm * 32 + mi * 16 + hsel * 8 + r;
                if (FULL || (cok && row < M)) {
                    float val = acc[mi][ni][r] + bv;
                    if (RELU) val = val > 0.f ? val : 0.f;
                    C[(long long)row * ldc + col] = val;
                }
            }
        }
    }
}

// ---------------------------------------------------------------------------
// Transpose QKV weights [H,D,DH] -> [D, H*DH] (one launch does Wq, Wk, Wv)
// ---------------------------------------------------------------------------
__global__ __launch_bounds__(256) void transpose_w_kernel(
    const float* __restrict__ w0, const float* __restrict__ w1,
    const float* __restrict__ w2, float* __restrict__ t0,
    float* __restrict__ t1, float* __restrict__ t2)
{
    const int o = blockIdx.x * 256 + threadIdx.x;     // output index
    if (o >= H_ * D_ * DH_) return;
    const int e = o & (DH_ - 1);
    const int r = o >> 6;
    const int h = r % H_;
    const int d = r / H_;
    const int in = h * (D_ * DH_) + d * DH_ + e;
    const float* w = (blockIdx.y == 0) ? w0 : (blockIdx.y == 1) ? w1 : w2;
    float*       t = (blockIdx.y == 0) ? t0 : (blockIdx.y == 1) ? t1 : t2;
    t[o] = w[in];
}

// ---------------------------------------------------------------------------
// h[b,t,d] = x[b,t]*w_pe[d] + b_pe[d] + pm[t,d]
// ---------------------------------------------------------------------------
__global__ __launch_bounds__(256) void pos_encode_kernel(
    const float* __restrict__ x, const float* __restrict__ w_pe,
    const float* __restrict__ b_pe, float* __restrict__ h)
{
    const long long idx = (long long)blockIdx.x * 256 + threadIdx.x;
    const long long total = (long long)B_ * T_ * D_;
    if (idx >= total) return;
    const int d = (int)(idx % D_);
    const long long bt = idx / D_;
    const int t = (int)(bt % T_);
    const int i = d >> 1;
    const float c = -2.0f * 9.2103403719762f / (float)D_;
    const float ang = (float)t * __expf((float)i * c);
    const float pmv = (d & 1) ? __cosf(ang) : __sinf(ang);
    h[idx] = x[bt] * w_pe[d] + b_pe[d] + pmv;
}

// ---------------------------------------------------------------------------
// Causal softmax over score rows: scores layout [H,B,T,T], one block per row.
// ---------------------------------------------------------------------------
__global__ __launch_bounds__(256) void softmax_kernel(float* __restrict__ scores)
{
    __shared__ float red[256];
    const long long rowIdx = blockIdx.x;
    const int t = (int)(rowIdx % T_);
    float* p = scores + rowIdx * (long long)T_;
    const int i = threadIdx.x;
    const float scale = 0.125f;  // 1/sqrt(DH)
    const float v = (i <= t) ? p[i] * scale : -3.4e38f;
    red[i] = v;
    __syncthreads();
    for (int s = 128; s > 0; s >>= 1) {
        if (i < s) red[i] = fmaxf(red[i], red[i + s]);
        __syncthreads();
    }
    const float m = red[0];
    __syncthreads();
    const float e = (i <= t) ? __expf(v - m) : 0.f;
    red[i] = e;
    __syncthreads();
    for (int s = 128; s > 0; s >>= 1) {
        if (i < s) red[i] += red[i + s];
        __syncthreads();
    }
    p[i] = e / red[0];
}

// ---------------------------------------------------------------------------
// LayerNorm over (T,D) per sample:  h[b] = LN(h[b] + res[b]) * gamma + beta
// ---------------------------------------------------------------------------
__global__ __launch_bounds__(1024) void layernorm_kernel(
    float* __restrict__ h, const float* __restrict__ res,
    const float* __restrict__ gamma, const float* __restrict__ beta)
{
    __shared__ float s1[1024];
    __shared__ float s2[1024];
    const long long N = (long long)T_ * D_;
    float* hp = h + (long long)blockIdx.x * N;
    const float* rp = res + (long long)blockIdx.x * N;

    float a1 = 0.f, a2 = 0.f;
    for (long long i = threadIdx.x; i < N; i += 1024) {
        const float v = hp[i] + rp[i];
        a1 += v;
        a2 += v * v;
    }
    s1[threadIdx.x] = a1;
    s2[threadIdx.x] = a2;
    __syncthreads();
    for (int s = 512; s > 0; s >>= 1) {
        if (threadIdx.x < s) {
            s1[threadIdx.x] += s1[threadIdx.x + s];
            s2[threadIdx.x] += s2[threadIdx.x + s];
        }
        __syncthreads();
    }
    const float invN = 1.f / (float)N;
    const float mu = s1[0] * invN;
    const float var = s2[0] * invN - mu * mu;
    const float rs = rsqrtf(var + 1e-5f);
    for (long long i = threadIdx.x; i < N; i += 1024) {
        const float v = hp[i] + rp[i];
        hp[i] = (v - mu) * rs * gamma[i] + beta[i];
    }
}

// ---------------------------------------------------------------------------
// Head split-K reduction: out[b,v] = bias[v] + sum_s part[s][b,v]
// ---------------------------------------------------------------------------
__global__ __launch_bounds__(256) void head_reduce_kernel(
    const float* __restrict__ part, const float* __restrict__ bias,
    float* __restrict__ out)
{
    const int idx = blockIdx.x * 256 + threadIdx.x;
    if (idx >= B_ * V_) return;
    float acc = bias[idx % V_];
#pragma unroll 4
    for (int s = 0; s < KSPLIT; ++s)
        acc += part[(long long)s * (B_ * V_) + idx];
    out[idx] = acc;
}

// ---------------------------------------------------------------------------
// Host-side launch
// ---------------------------------------------------------------------------
static inline dim3 gemm_grid(int N, int M, int Z) {
    return dim3((unsigned)((N + BN - 1) / BN), (unsigned)((M + BM - 1) / BM), (unsigned)Z);
}

extern "C" void kernel_launch(void* const* d_in, const int* in_sizes, int n_in,
                              void* d_out, int out_size, void* d_ws, size_t ws_size,
                              hipStream_t stream) {
    (void)in_sizes; (void)n_in; (void)out_size; (void)ws_size;

    const float* x      = (const float*)d_in[0];
    const float* w_pe   = (const float*)d_in[1];
    const float* b_pe   = (const float*)d_in[2];
    const float* Wq     = (const float*)d_in[3];
    const float* Wk     = (const float*)d_in[4];
    const float* Wv     = (const float*)d_in[5];
    const float* Wo     = (const float*)d_in[6];
    const float* ln1_w  = (const float*)d_in[7];
    const float* ln1_b  = (const float*)d_in[8];
    const float* ff1_w  = (const float*)d_in[9];
    const float* ff1_b  = (const float*)d_in[10];
    const float* ff2_w  = (const float*)d_in[11];
    const float* ff2_b  = (const float*)d_in[12];
    const float* ln2_w  = (const float*)d_in[13];
    const float* ln2_b  = (const float*)d_in[14];
    const float* head_w = (const float*)d_in[15];
    const float* head_b = (const float*)d_in[16];
    float* out = (float*)d_out;

    const long long BTD  = (long long)B_ * T_ * D_;        // 3,145,728
    const long long BIGN = (long long)H_ * B_ * T_ * T_;   // 12,582,912 (== B*T*F)
    const long long WSZ  = (long long)H_ * D_ * DH_;       // 589,824

    float* ws    = (float*)d_ws;
    float* h     = ws;                 // [B,T,D]
    float* tmp   = h     + BTD;        // [B,T,D]  residual branch (ctx@Wo / ff2)
    float* q     = tmp   + BTD;        // [B,T,H*DH]  (reused as ctx [B,T,D])
    float* kbuf  = q     + BTD;        // [B,T,H*DH]
    float* vbuf  = kbuf  + BTD;        // [B,T,H*DH]
    float* big   = vbuf  + BTD;        // scores [H,B,T,T] / ff activation [B,T,F]
    float* wqT   = big   + BIGN;       // [D, H*DH] transposed Wq
    float* wkT   = wqT   + WSZ;
    float* wvT   = wkT   + WSZ;
    float* hpart = wvT   + WSZ;        // [KSPLIT, B, V] head partials
    float* ctx   = q;                  // alias: q dead after scores

    const dim3 blk(256);
    const int BT = B_ * T_;

    // Positional encoding
    pos_encode_kernel<<<dim3((unsigned)((BTD + 255) / 256)), blk, 0, stream>>>(
        x, w_pe, b_pe, h);

    for (int l = 0; l < L_; ++l) {
        const float* Wql = Wq + (long long)l * WSZ;
        const float* Wkl = Wk + (long long)l * WSZ;
        const float* Wvl = Wv + (long long)l * WSZ;
        const float* Wol = Wo + (long long)l * D_ * D_;

        // Transpose Wq/Wk/Wv: [H,D,DH] -> [D, H*DH]
        transpose_w_kernel<<<dim3((unsigned)((WSZ + 255) / 256), 3), blk, 0, stream>>>(
            Wql, Wkl, Wvl, wqT, wkT, wvT);

        // Q/K/V: single FULL GEMM each  [B*T,768] @ [768,768] -> [B,T,H*DH]
        gemm_wmma_kernel<0,0,1,0><<<gemm_grid(D_, BT, 1), blk, 0, stream>>>(
            h, wqT, nullptr, q, BT, D_, D_, D_, D_, D_,
            0, 0, 0, 0, 0, 0, 1);
        gemm_wmma_kernel<0,0,1,0><<<gemm_grid(D_, BT, 1), blk, 0, stream>>>(
            h, wkT, nullptr, kbuf, BT, D_, D_, D_, D_, D_,
            0, 0, 0, 0, 0, 0, 1);
        gemm_wmma_kernel<0,0,1,0><<<gemm_grid(D_, BT, 1), blk, 0, stream>>>(
            h, wvT, nullptr, vbuf, BT, D_, D_, D_, D_, D_,
            0, 0, 0, 0, 0, 0, 1);

        // scores[h,b] = Q[h,b] @ K[h,b]^T  (M=N=256, K=64: FULL)
        // Q/K: [B,T,H*DH]: head stride 64, batch stride T*768, row stride 768
        const long long hS = (long long)DH_;
        const long long bS = (long long)T_ * D_;
        const long long sC1 = (long long)B_ * T_ * T_;
        const long long sC2 = (long long)T_ * T_;
        gemm_wmma_kernel<1,0,1,0><<<gemm_grid(T_, T_, H_ * B_), blk, 0, stream>>>(
            q, kbuf, nullptr, big, T_, T_, DH_, D_, D_, T_,
            hS, bS, hS, bS, sC1, sC2, B_);

        // causal softmax (scale fused)
        softmax_kernel<<<dim3((unsigned)(H_ * B_ * T_)), blk, 0, stream>>>(big);

        // ctx[b,t,h*DH+e] = attn[h,b] @ V[h,b]  (M=T, N=64: guarded)
        gemm_wmma_kernel<0,0,0,0><<<gemm_grid(DH_, T_, H_ * B_), blk, 0, stream>>>(
            big, vbuf, nullptr, ctx, T_, DH_, T_, T_, D_, D_,
            sC1, sC2, hS, bS, hS, bS, B_);

        // ctx @ Wo -> tmp   (M=4096, N=768, K=768: FULL)
        gemm_wmma_kernel<0,0,1,0><<<gemm_grid(D_, BT, 1), blk, 0, stream>>>(
            ctx, Wol, nullptr, tmp, BT, D_, D_, D_, D_, D_,
            0, 0, 0, 0, 0, 0, 1);

        // h = LN(h + tmp)
        layernorm_kernel<<<dim3(B_), dim3(1024), 0, stream>>>(
            h, tmp, ln1_w + (long long)l * T_ * D_, ln1_b + (long long)l * T_ * D_);

        // FFN: big = relu(h @ ff1_w + ff1_b)  (4096x3072x768: FULL+RELU)
        gemm_wmma_kernel<0,1,1,0><<<gemm_grid(F_, BT, 1), blk, 0, stream>>>(
            h, ff1_w + (long long)l * D_ * F_, ff1_b + (long long)l * F_, big,
            BT, F_, D_, D_, F_, F_, 0, 0, 0, 0, 0, 0, 1);
        //      tmp = big @ ff2_w + ff2_b      (4096x768x3072: FULL)
        gemm_wmma_kernel<0,0,1,0><<<gemm_grid(D_, BT, 1), blk, 0, stream>>>(
            big, ff2_w + (long long)l * F_ * D_, ff2_b + (long long)l * D_, tmp,
            BT, D_, F_, F_, D_, D_, 0, 0, 0, 0, 0, 0, 1);

        // h = LN(h + tmp)
        layernorm_kernel<<<dim3(B_), dim3(1024), 0, stream>>>(
            h, tmp, ln2_w + (long long)l * T_ * D_, ln2_b + (long long)l * T_ * D_);
    }

    // head, split-K: partials[s] = h[:, s*4096:(s+1)*4096] @ head_w[s*4096:...]
    // A: [16, 196608], B: [196608, 1024] streamed nontemporally (805 MB, used once)
    gemm_wmma_kernel<0,0,0,1><<<gemm_grid(V_, B_, KSPLIT), blk, 0, stream>>>(
        h, head_w, nullptr, hpart, B_, V_, KCHUNK, T_ * D_, V_, V_,
        (long long)KCHUNK, 0, (long long)KCHUNK * V_, 0,
        (long long)B_ * V_, 0, 1);

    // out = sum_s partials[s] + head_b   (fixed order -> deterministic)
    head_reduce_kernel<<<dim3((unsigned)((B_ * V_ + 255) / 256)), blk, 0, stream>>>(
        hpart, head_b, out);
}
